// Critic_47777216200758
// MI455X (gfx1250) — compile-verified, matrix operand order checked by reading, not run on hardware
//
#include <hip/hip_runtime.h>
#include <cstdint>

typedef __attribute__((ext_vector_type(16))) __bf16 v16bf;
typedef __attribute__((ext_vector_type(8)))  float  v8f;

namespace {

struct GSeg { const float* ptr; const int* idx; int width; int ld; int period; int shift; };
struct GDesc { GSeg s[3]; int nseg; };
struct MLPP {
  const unsigned* W0p; int nKt0; int K0;   // packed bf16 B-fragments, K0 padded to 32
  const unsigned* W1p; int nKt1;           // packed bf16 B-fragments, K1 = H
  const float *b0, *b1, *g, *beta;
};

__device__ __forceinline__ unsigned pack_bf(float a, float b) {
  unsigned short ha = __builtin_bit_cast(unsigned short, (__bf16)a);
  unsigned short hb = __builtin_bit_cast(unsigned short, (__bf16)b);
  return (unsigned)ha | ((unsigned)hb << 16);
}

__device__ __forceinline__ float siluf(float x) { return x / (1.0f + __expf(-x)); }

// Pack a row-major f32 weight W[K][N] into WMMA B-fragment layout:
// out[((ntile*nKt + kt)*32 + lane)*8 + d] = bf16x2 of B[k0][n], B[k0+1][n]
// Zero-pads k >= K, which lets the GEMM A-path read garbage in padded
// columns (product is zeroed by B).
__global__ void k_pack_w(const float* __restrict__ W, int K, int N, int nKt,
                         unsigned* __restrict__ out, int total) {
  int i = blockIdx.x * blockDim.x + threadIdx.x;
  if (i >= total) return;
  int d    = i & 7;
  int lane = (i >> 3) & 31;
  int tile = i >> 8;               // ntile * nKt + kt
  int kt = tile % nKt, ntile = tile / nKt;
  int n  = ntile * 16 + (lane & 15);
  int kb = (lane & 16) ? 8 : 0;
  int k0 = kt * 32 + kb + ((d >> 2) << 4) + ((d & 3) << 1);
  float a = (k0 < K)     ? W[(size_t)k0 * N + n]       : 0.0f;
  float b = (k0 + 1 < K) ? W[(size_t)(k0 + 1) * N + n] : 0.0f;
  out[i] = pack_bf(a, b);
}

// ---- GEMM 1: hid = silu(gather(A) @ W + bias) ----
// 64x64 tile, 8 waves. A gathered branchlessly: the segment pointer is
// selected ONCE per 8-column strip (all segment boundaries are multiples
// of 8, so strips never straddle), per-element work is just a clamp.
// A staged in LDS as packed bf16 pairs (b128 stores/loads); B fragments
// come straight from pre-packed weights (2x global_load_b128).
// Output written in packed bf16 A-fragment layout for the next GEMM.
__global__ __launch_bounds__(256) void k_gemm_silu(GDesc g,
    const unsigned* __restrict__ Wp, int nKt, int K, const float* __restrict__ bias,
    int M, int N, unsigned* __restrict__ outp) {
  __shared__ unsigned Asu[64][20];   // 16 bf16-pairs per row; stride 20 dw: conflict-free
  __shared__ float Cs[64][65];
  const int bm = blockIdx.x * 64, bn = blockIdx.y * 64;
  const int t = threadIdx.x, lane = t & 31, w = t >> 5;  // wave32
  const int mt = w & 3, nt0 = (w >> 2) << 1;

  // ---- hoisted gather setup: one A-tile row per thread, computed once ----
  const int r = bm + (t >> 2);            // this thread's A row
  const int cq = (t & 3) << 3;            // 8-column strip within the 32-wide k-slice
  const int rEff = (r < M) ? r : (M - 1); // clamp: garbage rows are never stored
  const int Kc = K - 1;
  const float *rp0, *rp1, *rp2;
  int en0, en1;
  {
    GSeg s0 = g.s[0];
    int rr0 = s0.idx ? (s0.idx[rEff % s0.period] + (rEff / s0.period) * s0.shift)
                     : (rEff % s0.period);
    rp0 = s0.ptr + (size_t)rr0 * s0.ld;
    en0 = s0.width;
    rp1 = rp0; rp2 = rp0; en1 = 0x40000000;
    if (g.nseg >= 2) {
      GSeg s1 = g.s[1];
      int rr1 = s1.idx ? (s1.idx[rEff % s1.period] + (rEff / s1.period) * s1.shift)
                       : (rEff % s1.period);
      rp1 = s1.ptr + (size_t)rr1 * s1.ld - en0;
      en1 = en0 + s1.width;
      rp2 = rp1;
      if (g.nseg == 3) {
        GSeg s2 = g.s[2];
        int rr2 = s2.idx ? (s2.idx[rEff % s2.period] + (rEff / s2.period) * s2.shift)
                         : (rEff % s2.period);
        rp2 = s2.ptr + (size_t)rr2 * s2.ld - en1;
      }
    }
  }

  v8f acc0 = {}, acc1 = {};
  const int rloc = t >> 2;
  const int ntA = (bn >> 4) + nt0;
  for (int kt = 0; kt < nKt; ++kt) {
    const int kk = kt * 32;
    {
      const int c0s = kk + cq;                 // strip start (multiple of 8)
      const int cS = (c0s < Kc) ? c0s : Kc;
      const float* p = rp2;                    // one select chain per strip
      p = (cS < en1) ? rp1 : p;
      p = (cS < en0) ? rp0 : p;
      unsigned tmp[4];
      #pragma unroll
      for (int jj = 0; jj < 4; ++jj) {
        int cA = c0s + (jj << 1), cB = cA + 1;
        cA = (cA < Kc) ? cA : Kc;              // cheap clamp; pad cols hit zeroed B
        cB = (cB < Kc) ? cB : Kc;
        tmp[jj] = pack_bf(p[cA], p[cB]);
      }
      *(uint4*)&Asu[rloc][cq >> 1] = *(uint4*)&tmp[0];   // one ds_store_b128
    }
    __syncthreads();
    union { v16bf v; unsigned u[8]; } fa, fb0, fb1;
    {
      int rr = mt * 16 + (lane & 15);
      int kb = (lane & 16) ? 4 : 0;             // pair units
      *(uint4*)&fa.u[0] = *(const uint4*)&Asu[rr][kb];       // ds_load_b128
      *(uint4*)&fa.u[4] = *(const uint4*)&Asu[rr][kb + 8];   // ds_load_b128
    }
    {
      const unsigned* pb0 = Wp + ((((size_t)ntA) * nKt + kt) * 32 + lane) * 8;
      const unsigned* pb1 = Wp + ((((size_t)(ntA + 1)) * nKt + kt) * 32 + lane) * 8;
      *(uint4*)&fb0.u[0] = *(const uint4*)pb0;
      *(uint4*)&fb0.u[4] = *(const uint4*)(pb0 + 4);
      *(uint4*)&fb1.u[0] = *(const uint4*)pb1;
      *(uint4*)&fb1.u[4] = *(const uint4*)(pb1 + 4);
      if (kt + 1 < nKt) __builtin_prefetch(pb0 + 256, 0, 1);
    }
    acc0 = __builtin_amdgcn_wmma_f32_16x16x32_bf16(false, fa.v, false, fb0.v, (short)0, acc0, false, false);
    acc1 = __builtin_amdgcn_wmma_f32_16x16x32_bf16(false, fa.v, false, fb1.v, (short)0, acc1, false, false);
    __syncthreads();
  }

  // stage (bias + silu) into LDS, then transpose-pack to A-fragment layout
  {
    const int rb = mt * 16 + ((lane & 16) ? 8 : 0);
    const int c0 = nt0 * 16 + (lane & 15);
    const float bv0 = bias[bn + c0], bv1 = bias[bn + c0 + 16];
    #pragma unroll
    for (int i = 0; i < 8; ++i) {
      Cs[rb + i][c0]      = siluf(acc0[i] + bv0);
      Cs[rb + i][c0 + 16] = siluf(acc1[i] + bv1);
    }
  }
  __syncthreads();
  {
    const int tt = t >> 5;            // 8 output tiles: 4 row-tiles x 2 k-tiles
    const int rt = tt >> 1, ktl = tt & 1;
    const int m = rt * 16 + (lane & 15);
    const int kb = (lane & 16) ? 8 : 0;
    const size_t rowTile = (size_t)(bm >> 4) + rt;
    const int kTileG = (bn >> 5) + ktl;
    unsigned* po = outp + ((rowTile * (N >> 5) + kTileG) * 32 + lane) * 8;
    unsigned tmp[8];
    #pragma unroll
    for (int i = 0; i < 8; ++i) {
      int k0 = ktl * 32 + kb + ((i >> 2) << 4) + ((i & 3) << 1);
      tmp[i] = pack_bf(Cs[m][k0], Cs[m][k0 + 1]);
    }
    *(uint4*)po = *(uint4*)&tmp[0];
    *(uint4*)(po + 4) = *(uint4*)&tmp[4];
  }
}

// ---- GEMM 2: out = [res +] LayerNorm(X @ W + bias) * gamma + beta ----
// BM=16, BN=N (whole row in block). A and B fragments come pre-packed from
// global memory: the K-loop has no LDS and no barriers at all.
__global__ __launch_bounds__(256) void k_gemm_ln(const unsigned* __restrict__ Xp, int nKt,
    const unsigned* __restrict__ Wp, const float* __restrict__ bias,
    const float* __restrict__ gamma, const float* __restrict__ beta,
    const float* __restrict__ res, int M, int N, int lgN, float* __restrict__ out) {
  __shared__ float C[16][513];
  __shared__ float redS[16][16], redQ[16][16];
  __shared__ float mu[16], rsg[16];
  const int t = threadIdx.x, lane = t & 31, w = t >> 5;
  const int tpw = N >> 7;            // 16-wide n-tiles per wave (8 waves)
  const int bm = blockIdx.x * 16;
  v8f acc[4] = {};
  for (int kt = 0; kt < nKt; ++kt) {
    union { v16bf v; unsigned u[8]; } fa;
    const unsigned* pa = Xp + (((size_t)blockIdx.x * nKt + kt) * 32 + lane) * 8;
    *(uint4*)&fa.u[0] = *(const uint4*)pa;
    *(uint4*)&fa.u[4] = *(const uint4*)(pa + 4);
    if (kt + 1 < nKt) __builtin_prefetch(pa + 256, 0, 1);
    #pragma unroll
    for (int tt = 0; tt < 4; ++tt) {
      if (tt < tpw) {
        int nt = w * tpw + tt;
        union { v16bf v; unsigned u[8]; } fb;
        const unsigned* pb = Wp + (((size_t)nt * nKt + kt) * 32 + lane) * 8;
        *(uint4*)&fb.u[0] = *(const uint4*)pb;
        *(uint4*)&fb.u[4] = *(const uint4*)(pb + 4);
        acc[tt] = __builtin_amdgcn_wmma_f32_16x16x32_bf16(false, fa.v, false, fb.v, (short)0, acc[tt], false, false);
      }
    }
  }
  const int rb = (lane & 16) ? 8 : 0;
  #pragma unroll
  for (int tt = 0; tt < 4; ++tt) {
    if (tt < tpw) {
      int n = (w * tpw + tt) * 16 + (lane & 15);
      float bv = bias[n];
      #pragma unroll
      for (int i = 0; i < 8; ++i) C[rb + i][n] = acc[tt][i] + bv;
    }
  }
  __syncthreads();
  {
    int r = t >> 4, p = t & 15, chunk = N >> 4;
    float s = 0.0f, q = 0.0f;
    for (int c = p * chunk; c < (p + 1) * chunk; ++c) { float v = C[r][c]; s += v; q += v * v; }
    redS[r][p] = s; redQ[r][p] = q;
  }
  __syncthreads();
  if (t < 16) {
    float s = 0.0f, q = 0.0f;
    #pragma unroll
    for (int p = 0; p < 16; ++p) { s += redS[t][p]; q += redQ[t][p]; }
    float m = s / (float)N;
    float var = q / (float)N - m * m;
    mu[t] = m; rsg[t] = rsqrtf(var + 1e-5f);
  }
  __syncthreads();
  const int nMask = N - 1;
  for (int i = t; i < 16 * N; i += 256) {
    int r = i >> lgN, n = i & nMask, row = bm + r;
    if (row < M) {
      float v = (C[r][n] - mu[r]) * rsg[r] * gamma[n] + beta[n];
      if (res) v += res[(size_t)row * N + n];
      out[(size_t)row * N + n] = v;
    }
  }
}

// dst edges are contiguous per (batch,node): dense strided reduction, no atomics
__global__ void k_segsum(const float* __restrict__ e, float* __restrict__ agg,
                         int nodes, int h, int lgH, int deg) {
  int i = blockIdx.x * blockDim.x + threadIdx.x;
  if (i >= nodes * h) return;
  int node = i >> lgH, c = i & (h - 1);
  const float* p = e + (size_t)node * deg * h + c;
  float s = 0.0f;
  for (int k = 0; k < deg; ++k) s += p[(size_t)k * h];
  agg[i] = s;
}

__global__ void k_add(const float* a, const float* b, float* c, int n) {
  int i = blockIdx.x * blockDim.x + threadIdx.x;
  if (i < n) c[i] = a[i] + b[i];
}

__global__ void k_mean(const float* __restrict__ h, float* __restrict__ out, int nn) {
  int i = blockIdx.x * blockDim.x + threadIdx.x;
  if (i >= 2 * 512) return;
  int b = i >> 9, c = i & 511;
  float s = 0.0f;
  for (int n = 0; n < nn; ++n) s += h[(size_t)(b * nn + n) * 512 + c];
  out[i] = s / (float)nn;
}

__global__ __launch_bounds__(256) void k_head(const float* __restrict__ m,
    const float* __restrict__ W1, const float* __restrict__ b1,
    const float* __restrict__ W2, float* __restrict__ out) {
  __shared__ float red[256];
  int b = blockIdx.x, t = threadIdx.x;
  float s = b1[t];
  for (int k = 0; k < 512; ++k) s += m[b * 512 + k] * W1[k * 256 + t];
  s = siluf(s);
  red[t] = s * W2[t];
  __syncthreads();
  for (int off = 128; off > 0; off >>= 1) {
    if (t < off) red[t] += red[t + off];
    __syncthreads();
  }
  if (t == 0) out[b] = red[0];
}

}  // namespace

extern "C" void kernel_launch(void* const* d_in, const int* in_sizes, int n_in,
                              void* d_out, int out_size, void* d_ws, size_t ws_size,
                              hipStream_t stream) {
  (void)n_in; (void)out_size; (void)ws_size;
  static const int HID[4]  = {128, 256, 256, 512};
  static const int LGH[4]  = {7, 8, 8, 9};
  static const int NSRC[4] = {40320, 10242, 2562, 642};
  static const int NDST[4] = {10242, 2562, 642, 162};
  static const int INS[4]  = {92, 128, 256, 256};
  const int NERA = 40320, BSX = 2;

  // Detect pytree flatten order: key-sorted (era_latlons first) vs insertion (x first).
  const bool sorted = (in_sizes[0] == NERA * 4);
  auto F = [&](int i) { return (const float*)d_in[i]; };
  auto I = [&](int i) { return (const int*)d_in[i]; };

  const float* x = nullptr; const float* era = nullptr;
  const float* fW1 = nullptr; const float* fb1 = nullptr; const float* fW2 = nullptr;
  int lvlBase[4], mapBase[4], procBase[4];
  if (sorted) {
    era = F(0); x = F(280);
    fW1 = F(37); fW2 = F(38); fb1 = F(39);
    for (int L = 0; L < 4; ++L) { lvlBase[L] = 1 + 9 * L; mapBase[L] = 40 + 30 * L; procBase[L] = 160 + 30 * L; }
  } else {
    x = F(0); era = F(1);
    fW1 = F(278); fb1 = F(279); fW2 = F(280);
    for (int L = 0; L < 4; ++L) { lvlBase[L] = 2 + 9 * L; mapBase[L] = 38 + 30 * L; procBase[L] = 158 + 30 * L; }
  }

  // Workspace carve.
  char* wsb = (char*)d_ws;
  size_t cur = 0;
  auto carveB = [&](size_t bytes) {
    void* p = (void*)(wsb + cur);
    cur = (cur + bytes + 255) & ~(size_t)255;
    return p;
  };
  float* xs    = (float*)carveB(10321920u * 4);  // BS*n_src*h max (level 0)
  float* xd    = (float*)carveB(2621952u * 4);
  float* xp    = (float*)carveB(2621952u * 4);
  float* hsrc  = (float*)carveB(2621952u * 4);
  float* emap  = (float*)carveB(31463424u * 4);
  float* eproc = (float*)carveB(15731712u * 4);
  float* agg   = (float*)carveB(2621952u * 4);
  float* mout  = (float*)carveB(1024u * 4);
  unsigned* hidp = (unsigned*)carveB(15731712u * 4);  // packed bf16 hidden (A-frag layout)

  // Prepack each MLP's weights into WMMA B-fragment layout (bf16), once per launch.
  auto prep = [&](int b, int K0, int H) {
    MLPP m;
    int nKt0 = (K0 + 31) / 32, nKt1 = H / 32;
    int tot0 = (H / 16) * nKt0 * 256;
    int tot1 = (H / 16) * nKt1 * 256;
    unsigned* w0 = (unsigned*)carveB((size_t)tot0 * 4);
    unsigned* w1 = (unsigned*)carveB((size_t)tot1 * 4);
    k_pack_w<<<(tot0 + 255) / 256, 256, 0, stream>>>(F(b), K0, H, nKt0, w0, tot0);
    k_pack_w<<<(tot1 + 255) / 256, 256, 0, stream>>>(F(b + 1), H, H, nKt1, w1, tot1);
    m.W0p = w0; m.nKt0 = nKt0; m.K0 = K0; m.W1p = w1; m.nKt1 = nKt1;
    m.b0 = F(b + 2); m.b1 = F(b + 3);
    if (sorted) { m.beta = F(b + 4); m.g = F(b + 5); }  // sorted keys: W,b,beta,g
    else        { m.g = F(b + 4); m.beta = F(b + 5); }  // insertion:   W,b,g,beta
    return m;
  };

  auto runMLP = [&](const GDesc& g, int M, int H, int lgH, const MLPP& p,
                    float* outF, const float* res) {
    dim3 g1((unsigned)((M + 63) / 64), (unsigned)(H / 64));
    k_gemm_silu<<<g1, 256, 0, stream>>>(g, p.W0p, p.nKt0, p.K0, p.b0, M, H, hidp);
    dim3 g2((unsigned)((M + 15) / 16));
    k_gemm_ln<<<g2, 256, 0, stream>>>(hidp, p.nKt1, p.W1p, p.b1, p.g, p.beta, res, M, H, lgH, outF);
  };

  for (int L = 0; L < 4; ++L) {
    const int h = HID[L], lgh = LGH[L], nsrc = NSRC[L], ndst = NDST[L];
    const int em = 12 * ndst, ep = 6 * ndst;
    const int Ns = BSX * nsrc, Nn = BSX * ndst, EM = BSX * em, EP = BSX * ep;
    const int lb = lvlBase[L];
    const int *msrc, *mdst, *psrc, *pdst;
    const float *mattr, *pattr, *latl;
    if (sorted) {
      latl = F(lb + 0); mattr = F(lb + 1); mdst = I(lb + 2); msrc = I(lb + 3);
      pattr = F(lb + 6); pdst = I(lb + 7); psrc = I(lb + 8);
    } else {
      msrc = I(lb + 2); mdst = I(lb + 3); mattr = F(lb + 4);
      psrc = I(lb + 5); pdst = I(lb + 6); pattr = F(lb + 7); latl = F(lb + 8);
    }
    MLPP mSrc, mDst, mEdge, mBlkE, mBlkN, pEdge, pBE[2], pBN[2];
    const int mb = mapBase[L], pb = procBase[L];
    if (sorted) {  // mapper keys: blocks < dst_emb < edge_emb < src_emb
      mBlkE = prep(mb + 0, 3 * h, h); mBlkN = prep(mb + 6, 2 * h, h);
      mDst  = prep(mb + 12, 4, h);    mEdge = prep(mb + 18, 3, h);  mSrc = prep(mb + 24, INS[L], h);
      pBE[0] = prep(pb + 0, 3 * h, h);  pBN[0] = prep(pb + 6, 2 * h, h);
      pBE[1] = prep(pb + 12, 3 * h, h); pBN[1] = prep(pb + 18, 2 * h, h);
      pEdge  = prep(pb + 24, 3, h);
    } else {       // insertion: src_emb, dst_emb, edge_emb, blocks / edge_emb, blocks
      mSrc = prep(mb + 0, INS[L], h); mDst = prep(mb + 6, 4, h); mEdge = prep(mb + 12, 3, h);
      mBlkE = prep(mb + 18, 3 * h, h); mBlkN = prep(mb + 24, 2 * h, h);
      pEdge = prep(pb + 0, 3, h);
      pBE[0] = prep(pb + 6, 3 * h, h);  pBN[0] = prep(pb + 12, 2 * h, h);
      pBE[1] = prep(pb + 18, 3 * h, h); pBN[1] = prep(pb + 24, 2 * h, h);
    }

    // 1. src_emb -> xs  (level 0: fused concat [x | tiled era_latlons])
    GDesc gS{};
    if (L == 0) {
      gS.nseg = 2;
      gS.s[0] = GSeg{x, nullptr, 88, 88, Ns, 0};
      gS.s[1] = GSeg{era, nullptr, 4, 4, NERA, 0};
    } else {
      gS.nseg = 1;
      gS.s[0] = GSeg{hsrc, nullptr, INS[L], INS[L], Ns, 0};
    }
    runMLP(gS, Ns, h, lgh, mSrc, xs, nullptr);

    // 2. dst_emb(latlons tiled over batch) -> xd
    GDesc gD{}; gD.nseg = 1; gD.s[0] = GSeg{latl, nullptr, 4, 4, ndst, 0};
    runMLP(gD, Nn, h, lgh, mDst, xd, nullptr);

    // 3. mapper edge_emb -> emap
    GDesc gE{}; gE.nseg = 1; gE.s[0] = GSeg{mattr, nullptr, 3, 3, em, 0};
    runMLP(gE, EM, h, lgh, mEdge, emap, nullptr);

    // 4. mapper block: e += MLP([xs[src], xd[dst], e]); agg; xd += MLP([xd, agg])
    GDesc gBE{}; gBE.nseg = 3;
    gBE.s[0] = GSeg{xs, msrc, h, h, em, nsrc};
    gBE.s[1] = GSeg{xd, mdst, h, h, em, ndst};
    gBE.s[2] = GSeg{emap, nullptr, h, h, EM, 0};
    runMLP(gBE, EM, h, lgh, mBlkE, emap, emap);
    k_segsum<<<(Nn * h + 255) / 256, 256, 0, stream>>>(emap, agg, Nn, h, lgh, 12);
    GDesc gBN{}; gBN.nseg = 2;
    gBN.s[0] = GSeg{xd, nullptr, h, h, Nn, 0};
    gBN.s[1] = GSeg{agg, nullptr, h, h, Nn, 0};
    runMLP(gBN, Nn, h, lgh, mBlkN, xd, xd);

    // 5. processor works on a copy (outer residual needs original xd)
    hipMemcpyAsync(xp, xd, (size_t)Nn * h * sizeof(float), hipMemcpyDeviceToDevice, stream);

    // 6. processor edge_emb -> eproc
    GDesc gPE{}; gPE.nseg = 1; gPE.s[0] = GSeg{pattr, nullptr, 3, 3, ep, 0};
    runMLP(gPE, EP, h, lgh, pEdge, eproc, nullptr);

    // 7. two processor blocks
    for (int blk = 0; blk < 2; ++blk) {
      GDesc ge{}; ge.nseg = 3;
      ge.s[0] = GSeg{xp, psrc, h, h, ep, ndst};
      ge.s[1] = GSeg{xp, pdst, h, h, ep, ndst};
      ge.s[2] = GSeg{eproc, nullptr, h, h, EP, 0};
      runMLP(ge, EP, h, lgh, pBE[blk], eproc, eproc);
      k_segsum<<<(Nn * h + 255) / 256, 256, 0, stream>>>(eproc, agg, Nn, h, lgh, 6);
      GDesc gn{}; gn.nseg = 2;
      gn.s[0] = GSeg{xp, nullptr, h, h, Nn, 0};
      gn.s[1] = GSeg{agg, nullptr, h, h, Nn, 0};
      runMLP(gn, Nn, h, lgh, pBN[blk], xp, xp);
    }

    // 8. h_next = xd + processed
    k_add<<<(Nn * h + 255) / 256, 256, 0, stream>>>(xd, xp, hsrc, Nn * h);
  }

  // mean over 162 nodes per batch, then final head -> d_out (BS x 1, f32)
  k_mean<<<(2 * 512 + 255) / 256, 256, 0, stream>>>(hsrc, mout, 162);
  k_head<<<2, 256, 0, stream>>>(mout, fW1, fb1, fW2, (float*)d_out);
}
